// GCNLayer_3006477107661
// MI455X (gfx1250) — matrix operand order repaired and checked by reading, not compiled, not run on hardware
//
#include <hip/hip_runtime.h>

#define DFEAT 128
#define WT_STRIDE 136  // padded LDS stride (halfs): 16-byte aligned rows, reduced bank conflicts

typedef __attribute__((ext_vector_type(16))) _Float16 v16h;
typedef __attribute__((ext_vector_type(8)))  _Float16 v8h;
typedef __attribute__((ext_vector_type(8)))  float    v8f;

// ---------------- utility: zero fill ----------------
__global__ void gcn_zero_f32(float* __restrict__ p, long n) {
    long i = (long)blockIdx.x * blockDim.x + threadIdx.x;
    long stride = (long)gridDim.x * blockDim.x;
    for (; i < n; i += stride) p[i] = 0.0f;
}

// ---------------- degree (in-degree over col) ----------------
__global__ void gcn_degree(const int* __restrict__ col, float* __restrict__ deg, int n_edges) {
    int e = blockIdx.x * blockDim.x + threadIdx.x;
    if (e < n_edges) atomicAdd(&deg[col[e]], 1.0f);
}

__global__ void gcn_deg_inv_sqrt(const float* __restrict__ deg, float* __restrict__ dis, int n_nodes) {
    int i = blockIdx.x * blockDim.x + threadIdx.x;
    if (i < n_nodes) {
        float d = deg[i];
        dis[i] = (d > 0.0f) ? rsqrtf(fmaxf(d, 1.0f)) : 0.0f;
    }
}

// ---------------- GEMM: h = x @ W via v_wmma_f32_16x16x32_f16 ----------------
// Block: 256 threads = 8 waves; computes 64 rows x 128 cols.
// Wave w: row-tile (w>>1), column tiles [(w&1)*4, (w&1)*4+4).
// W staged transposed in LDS as f16: Wt[n*WT_STRIDE + k] = W[k][n].
__global__ __launch_bounds__(256) void gcn_gemm_wmma(const float* __restrict__ x,
                                                     const float* __restrict__ W,
                                                     float* __restrict__ h,
                                                     int n_rows) {
    __shared__ _Float16 Wt[DFEAT * WT_STRIDE];
    for (int idx = threadIdx.x; idx < DFEAT * DFEAT; idx += 256) {
        int k = idx >> 7;       // W row
        int n = idx & 127;      // W col
        Wt[n * WT_STRIDE + k] = (_Float16)W[idx];   // coalesced read, transposed write
    }
    __syncthreads();

    const int wave    = threadIdx.x >> 5;
    const int lane    = threadIdx.x & 31;
    const int m       = lane & 15;   // matrix row (A) / col (B,C)
    const int grp     = lane >> 4;   // lane-group 0/1
    const int rowTile = wave >> 1;
    const int colHalf = wave & 1;
    const int rowBase = blockIdx.x * 64 + rowTile * 16;

    // clamp A-row address for tail block (row i of C only reads row i of A,
    // so garbage rows are harmless; stores are guarded below). EXEC stays all-1s.
    const int arow = rowBase + m;
    const int arowc = (arow < n_rows) ? arow : (n_rows - 1);
    const float* __restrict__ xr = x + (size_t)arowc * DFEAT;

    v8f acc[4] = {};
    #pragma unroll
    for (int k0 = 0; k0 < DFEAT; k0 += 32) {
        // A fragment (16x32 f16): lane grp g holds K = k0+16g .. +15
        v16h a;
        const float* __restrict__ xk = xr + k0 + 16 * grp;
        #pragma unroll
        for (int i = 0; i < 8; ++i) a[i] = (_Float16)xk[i];
        #pragma unroll
        for (int i = 0; i < 8; ++i) a[i + 8] = (_Float16)xk[i + 8];

        #pragma unroll
        for (int t = 0; t < 4; ++t) {
            const int cb = (colHalf * 4 + t) * 16;
            // B fragment (32x16 f16): lane = column cb+m, halfs = K = k0+16g .. +15
            const _Float16* __restrict__ wp = &Wt[(cb + m) * WT_STRIDE + k0 + 16 * grp];
            v8h b0 = *(const v8h*)wp;        // 16B aligned (WT_STRIDE*2 = 272 = 17*16)
            v8h b1 = *(const v8h*)(wp + 8);
            v16h b;
            #pragma unroll
            for (int i = 0; i < 8; ++i) { b[i] = b0[i]; b[i + 8] = b1[i]; }

            acc[t] = __builtin_amdgcn_wmma_f32_16x16x32_f16(
                false, a, false, b, (short)0, acc[t], false, false);
        }
    }

    // C/D layout: VGPR r holds row (r + 8*grp), col = m
    #pragma unroll
    for (int t = 0; t < 4; ++t) {
        const int cb = (colHalf * 4 + t) * 16;
        #pragma unroll
        for (int r = 0; r < 8; ++r) {
            int rr = rowBase + 8 * grp + r;
            if (rr < n_rows) h[(size_t)rr * DFEAT + cb + m] = acc[t][r];
        }
    }
}

// ---------------- edge scatter: agg[col] += h[row] * norm ----------------
// One wave per edge: float4 coalesced gather of a 512B row, 4 f32 atomics/lane.
__global__ void gcn_scatter(const int* __restrict__ row_idx, const int* __restrict__ col_idx,
                            const float* __restrict__ h, const float* __restrict__ dis,
                            float* __restrict__ agg, int n_edges) {
    int gtid = blockIdx.x * blockDim.x + threadIdx.x;
    int wid  = gtid >> 5;
    int lane = threadIdx.x & 31;
    if (wid >= n_edges) return;
    int r = row_idx[wid];
    int c = col_idx[wid];
    float norm = dis[r] * dis[c];
    const float4 hv = ((const float4*)(h + (size_t)r * DFEAT))[lane];
    float* ap = agg + (size_t)c * DFEAT + lane * 4;
    atomicAdd(ap + 0, hv.x * norm);
    atomicAdd(ap + 1, hv.y * norm);
    atomicAdd(ap + 2, hv.z * norm);
    atomicAdd(ap + 3, hv.w * norm);
}

// ---------------- BN statistics over (agg + bias) ----------------
__global__ void gcn_stats(const float* __restrict__ agg, const float* __restrict__ bias,
                          float* __restrict__ sum, float* __restrict__ sumsq, int n_nodes) {
    int f = threadIdx.x;              // 128 threads = 128 features
    float bf = bias[f];
    float s = 0.0f, s2 = 0.0f;
    for (int node = blockIdx.x; node < n_nodes; node += gridDim.x) {
        float v = agg[(size_t)node * DFEAT + f] + bf;
        s  += v;
        s2 += v * v;
    }
    atomicAdd(&sum[f], s);
    atomicAdd(&sumsq[f], s2);
}

__global__ void gcn_finalize_stats(const float* __restrict__ sum, const float* __restrict__ sumsq,
                                   const float* __restrict__ gamma, const float* __restrict__ beta,
                                   float* __restrict__ scale, float* __restrict__ shift, int n_nodes) {
    int f = threadIdx.x;
    float inv_n = 1.0f / (float)n_nodes;
    float mean = sum[f] * inv_n;
    float var  = sumsq[f] * inv_n - mean * mean;   // biased variance
    float inv  = rsqrtf(var + 1e-5f);
    float sc   = gamma[f] * inv;
    scale[f] = sc;
    shift[f] = beta[f] - mean * sc;
}

// ---------------- apply: out = relu((agg+b)*scale + shift) + x (in place) ----------------
__global__ void gcn_apply(float* __restrict__ agg_out, const float* __restrict__ x,
                          const float* __restrict__ bias, const float* __restrict__ scale,
                          const float* __restrict__ shift, long n4) {
    long i = (long)blockIdx.x * blockDim.x + threadIdx.x;
    if (i >= n4) return;
    int fb = (int)((i & 31) * 4);     // 32 float4 per 128-feature row
    float4 v  = ((const float4*)agg_out)[i];
    float4 xv = ((const float4*)x)[i];
    float4 o;
    o.x = fmaxf((v.x + bias[fb + 0]) * scale[fb + 0] + shift[fb + 0], 0.0f) + xv.x;
    o.y = fmaxf((v.y + bias[fb + 1]) * scale[fb + 1] + shift[fb + 1], 0.0f) + xv.y;
    o.z = fmaxf((v.z + bias[fb + 2]) * scale[fb + 2] + shift[fb + 2], 0.0f) + xv.z;
    o.w = fmaxf((v.w + bias[fb + 3]) * scale[fb + 3] + shift[fb + 3], 0.0f) + xv.w;
    ((float4*)agg_out)[i] = o;
}

extern "C" void kernel_launch(void* const* d_in, const int* in_sizes, int n_in,
                              void* d_out, int out_size, void* d_ws, size_t ws_size,
                              hipStream_t stream) {
    const float* x     = (const float*)d_in[0];
    const int*   ei    = (const int*)d_in[1];     // [2, E] edge index (int32)
    const float* W     = (const float*)d_in[2];
    const float* bias  = (const float*)d_in[3];
    const float* gamma = (const float*)d_in[4];
    const float* beta  = (const float*)d_in[5];
    float* out = (float*)d_out;                    // doubles as agg accumulator

    const int n_nodes = in_sizes[0] / DFEAT;
    const int n_edges = in_sizes[1] / 2;
    const long hn = (long)n_nodes * DFEAT;

    const int* row_idx = ei;
    const int* col_idx = ei + n_edges;

    // workspace layout (floats)
    float* ws    = (float*)d_ws;
    float* h     = ws;                 // n_nodes*128
    float* deg   = h + hn;             // n_nodes
    float* dis   = deg + n_nodes;      // n_nodes
    float* sum   = dis + n_nodes;      // 128
    float* sumsq = sum + DFEAT;        // 128 (contiguous with sum)
    float* scale = sumsq + DFEAT;      // 128
    float* shift = scale + DFEAT;      // 128

    const int T = 256;

    // zero accumulators (d_out used as agg; deg; sum+sumsq)
    gcn_zero_f32<<<2048, T, 0, stream>>>(out, hn);
    gcn_zero_f32<<<128, T, 0, stream>>>(deg, (long)n_nodes);
    gcn_zero_f32<<<1, T, 0, stream>>>(sum, 2L * DFEAT);

    // degree + normalization factors
    gcn_degree<<<(n_edges + T - 1) / T, T, 0, stream>>>(col_idx, deg, n_edges);
    gcn_deg_inv_sqrt<<<(n_nodes + T - 1) / T, T, 0, stream>>>(deg, dis, n_nodes);

    // h = x @ W  (WMMA f16 -> f32)
    gcn_gemm_wmma<<<(n_nodes + 63) / 64, 256, 0, stream>>>(x, W, h, n_nodes);

    // edge scatter-add (one wave per edge)
    long scatter_threads = (long)n_edges * 32;
    gcn_scatter<<<(int)((scatter_threads + T - 1) / T), T, 0, stream>>>(
        row_idx, col_idx, h, dis, out, n_edges);

    // batch-norm stats, finalize, apply (+bias, relu, residual) in place
    gcn_stats<<<512, DFEAT, 0, stream>>>(out, bias, sum, sumsq, n_nodes);
    gcn_finalize_stats<<<1, DFEAT, 0, stream>>>(sum, sumsq, gamma, beta, scale, shift, n_nodes);
    gcn_apply<<<(int)((hn / 4 + T - 1) / T), T, 0, stream>>>(out, x, bias, scale, shift, hn / 4);
}
